// TransformerEncoder_77945066488218
// MI455X (gfx1250) — matrix operand order repaired and checked by reading, not compiled, not run on hardware
//
#include <hip/hip_runtime.h>

typedef __bf16 bf16;
typedef __attribute__((ext_vector_type(16))) __bf16 v16bf;
typedef __attribute__((ext_vector_type(8)))  float  v8f;

#define B_    4
#define L_    2048
#define D_    1024
#define H_    8
#define DH_   128
#define FF_   2048
#define M_TOT (B_ * L_)   // 8192 tokens
#define NHEAD (B_ * H_)   // 32 "heads" (raw reshape groups)

constexpr int BM = 128, BN = 128, BK = 64;
constexpr int LDSK = 72;  // padded LDS row stride (elems) -> conflict-free frag loads

// ---------------------------------------------------------------------------
// CDNA5 async global->LDS copy (ASYNCcnt-tracked, no VGPR staging)
// ---------------------------------------------------------------------------
__device__ __forceinline__ void async_ld16(unsigned lds_byte_off, const void* gaddr) {
    asm volatile("global_load_async_to_lds_b128 %0, %1, off"
                 :: "v"(lds_byte_off), "v"(gaddr) : "memory");
}
__device__ __forceinline__ void wait_async0() {
    asm volatile("s_wait_asynccnt 0x0" ::: "memory");
}

// ---------------------------------------------------------------------------
// f32 -> bf16 pack
// ---------------------------------------------------------------------------
__global__ void f32_to_bf16(const float* __restrict__ s, bf16* __restrict__ d, int n) {
    int i = blockIdx.x * 256 + threadIdx.x;
    if (i < n) d[i] = (bf16)s[i];
}

// ---------------------------------------------------------------------------
// C = alpha * (A @ B^T) + bias ; A: MxK bf16 row-major, B: NxK bf16 row-major
// optional relu, optional f32 and/or bf16 outputs, batched via blockIdx.z.
// Async double-buffered LDS tiles: DMA of tile k+1 overlaps WMMA of tile k.
// All of M, N divisible by 128 and K by 64 (true for every use here).
// ---------------------------------------------------------------------------
__global__ __launch_bounds__(256) void gemm_bt_kernel(
    const bf16* __restrict__ A, long long strideA, int lda,
    const bf16* __restrict__ Bm, long long strideB, int ldb,
    int K, const float* __restrict__ bias, float alpha, int relu,
    float* __restrict__ C32, bf16* __restrict__ C16,
    long long strideC, int ldc)
{
    __shared__ bf16 sA[2][BM * LDSK];
    __shared__ bf16 sB[2][BN * LDSK];

    const int tid  = threadIdx.x;
    const int lane = tid & 31;
    const int wave = tid >> 5;
    const int wm   = wave >> 2;   // 0..1 : 64 rows each
    const int wn   = wave & 3;    // 0..3 : 32 cols each
    const int m0   = blockIdx.y * BM;
    const int n0   = blockIdx.x * BN;
    const long long aB = (long long)blockIdx.z * strideA;
    const long long bB = (long long)blockIdx.z * strideB;
    const long long cB = (long long)blockIdx.z * strideC;

    // LDS byte offsets of the tile buffers (low 32 bits of flat LDS address)
    const unsigned sAoff = (unsigned)(unsigned long long)(void*)&sA[0][0];
    const unsigned sBoff = (unsigned)(unsigned long long)(void*)&sB[0][0];
    constexpr unsigned BUFB = (unsigned)(BM * LDSK * 2);  // bytes per buffer

    // each thread DMAs 4 x 16B for A and for B per tile
    const int fr   = tid >> 3;           // 0..31 base row
    const int fcol = (tid & 7) * 8;      // 0..56
    const unsigned ldsoff = (unsigned)((fr * LDSK + fcol) * 2);

    auto fill_async = [&](int buf, int k0) {
#pragma unroll
        for (int c = 0; c < 4; ++c) {
            int r = fr + c * 32;
            unsigned lo = ldsoff + (unsigned)(c * 32 * LDSK * 2) + (unsigned)buf * BUFB;
            async_ld16(sAoff + lo, A + aB + (long long)(m0 + r) * lda + k0 + fcol);
            async_ld16(sBoff + lo, Bm + bB + (long long)(n0 + r) * ldb + k0 + fcol);
        }
    };

    v8f acc[4][2] = {};

    fill_async(0, 0);
    wait_async0();
    __syncthreads();

    const int nK = K / BK;
    for (int kt = 0; kt < nK; ++kt) {
        const int cur = kt & 1;
        if (kt + 1 < nK) fill_async(cur ^ 1, (kt + 1) * BK);  // overlap with compute

#pragma unroll
        for (int ks = 0; ks < BK; ks += 32) {
            const int lh = lane >> 4;
            const int lm = lane & 15;
            union { v16bf v; int4 q[2]; } aF[4], bF[2];
            // A frag (16x32 bf16): lanes<16 hold K 0-7,16-23; lanes>=16 hold 8-15,24-31
#pragma unroll
            for (int mi = 0; mi < 4; ++mi) {
                const bf16* p = &sA[cur][(wm * 64 + mi * 16 + lm) * LDSK + ks + lh * 8];
                aF[mi].q[0] = *(const int4*)p;
                aF[mi].q[1] = *(const int4*)(p + 16);
            }
            // B frag (32x16 bf16): lanes<16 hold K 0-7,16-23 of col n; lanes>=16 hold 8-15,24-31
#pragma unroll
            for (int ni = 0; ni < 2; ++ni) {
                const bf16* p = &sB[cur][(wn * 32 + ni * 16 + lm) * LDSK + ks + lh * 8];
                bF[ni].q[0] = *(const int4*)p;
                bF[ni].q[1] = *(const int4*)(p + 16);
            }
#pragma unroll
            for (int mi = 0; mi < 4; ++mi)
#pragma unroll
                for (int ni = 0; ni < 2; ++ni)
                    acc[mi][ni] = __builtin_amdgcn_wmma_f32_16x16x32_bf16(
                        false, aF[mi].v, false, bF[ni].v,
                        (short)0, acc[mi][ni], false, false);
        }
        wait_async0();     // next tile fully landed in LDS (per-wave)
        __syncthreads();   // publish to all waves; also protects buffer reuse
    }

    const int lh = lane >> 4;
    const int lm = lane & 15;
#pragma unroll
    for (int mi = 0; mi < 4; ++mi)
#pragma unroll
        for (int ni = 0; ni < 2; ++ni) {
            int row0 = m0 + wm * 64 + mi * 16 + lh * 8;
            int col  = n0 + wn * 32 + ni * 16 + lm;
            float bv = bias ? bias[col] : 0.f;
#pragma unroll
            for (int r = 0; r < 8; ++r) {
                float v = acc[mi][ni][r] * alpha + bv;
                if (relu) v = fmaxf(v, 0.f);
                long long idx = cB + (long long)(row0 + r) * ldc + col;
                if (C32) C32[idx] = v;
                if (C16) C16[idx] = (bf16)v;
            }
        }
}

// ---------------------------------------------------------------------------
// Context GEMM: Ctx = P @ V ; P: L x L f32 (softmaxed atten), V: L x 128 bf16
// P converted to bf16 while filling LDS; V transposed into LDS [n][k].
// (Needs data transformation, so VGPR staging instead of async DMA.)
// ---------------------------------------------------------------------------
__global__ __launch_bounds__(256) void gemm_ctx_kernel(
    const float* __restrict__ P, long long strideP,
    const bf16* __restrict__ V, long long strideV,
    bf16* __restrict__ C16, long long strideC, int K)
{
    __shared__ bf16 sA[BM * LDSK];
    __shared__ bf16 sB[BN * LDSK];

    const int tid  = threadIdx.x;
    const int lane = tid & 31;
    const int wave = tid >> 5;
    const int wm   = wave >> 2;
    const int wn   = wave & 3;
    const int m0   = blockIdx.y * BM;            // n0 == 0 (BN == N == 128)
    const long long pB = (long long)blockIdx.z * strideP;
    const long long vB = (long long)blockIdx.z * strideV;
    const long long cB = (long long)blockIdx.z * strideC;

    v8f acc[4][2] = {};

    for (int k0 = 0; k0 < K; k0 += BK) {
        // A tile: 128x64 f32 -> bf16
#pragma unroll
        for (int c = 0; c < 8; ++c) {
            int q   = c * 256 + tid;
            int r   = q >> 4;           // 0..127
            int col = (q & 15) * 4;     // 0..60
            float4 f = *(const float4*)(P + pB + (long long)(m0 + r) * K + k0 + col);
            union { bf16 h[4]; uint2 u; } cv;
            cv.h[0] = (bf16)f.x; cv.h[1] = (bf16)f.y;
            cv.h[2] = (bf16)f.z; cv.h[3] = (bf16)f.w;
            *(uint2*)(&sA[r * LDSK + col]) = cv.u;
        }
        // B tile: V[k0..k0+63][0..127] row-major -> sB[n][k]
        {
            int k  = tid >> 2;          // 0..63
            int nb = (tid & 3) * 32;    // 0,32,64,96
            union { int4 q[4]; bf16 h[32]; } buf;
            const bf16* vp = V + vB + (long long)(k0 + k) * 128 + nb;
            buf.q[0] = ((const int4*)vp)[0];
            buf.q[1] = ((const int4*)vp)[1];
            buf.q[2] = ((const int4*)vp)[2];
            buf.q[3] = ((const int4*)vp)[3];
#pragma unroll
            for (int j = 0; j < 32; ++j) sB[(nb + j) * LDSK + k] = buf.h[j];
        }
        __syncthreads();

#pragma unroll
        for (int ks = 0; ks < BK; ks += 32) {
            const int lh = lane >> 4;
            const int lm = lane & 15;
            union { v16bf v; int4 q[2]; } aF[4], bF[2];
#pragma unroll
            for (int mi = 0; mi < 4; ++mi) {
                const bf16* p = &sA[(wm * 64 + mi * 16 + lm) * LDSK + ks + lh * 8];
                aF[mi].q[0] = *(const int4*)p;
                aF[mi].q[1] = *(const int4*)(p + 16);
            }
#pragma unroll
            for (int ni = 0; ni < 2; ++ni) {
                const bf16* p = &sB[(wn * 32 + ni * 16 + lm) * LDSK + ks + lh * 8];
                bF[ni].q[0] = *(const int4*)p;
                bF[ni].q[1] = *(const int4*)(p + 16);
            }
#pragma unroll
            for (int mi = 0; mi < 4; ++mi)
#pragma unroll
                for (int ni = 0; ni < 2; ++ni)
                    acc[mi][ni] = __builtin_amdgcn_wmma_f32_16x16x32_bf16(
                        false, aF[mi].v, false, bF[ni].v,
                        (short)0, acc[mi][ni], false, false);
        }
        __syncthreads();
    }

    const int lh = lane >> 4;
    const int lm = lane & 15;
#pragma unroll
    for (int mi = 0; mi < 4; ++mi)
#pragma unroll
        for (int ni = 0; ni < 2; ++ni) {
            int row0 = m0 + wm * 64 + mi * 16 + lh * 8;
            int col  = wn * 32 + ni * 16 + lm;
#pragma unroll
            for (int r = 0; r < 8; ++r)
                C16[cB + (long long)(row0 + r) * 128 + col] = (bf16)acc[mi][ni][r];
        }
}

// ---------------------------------------------------------------------------
// Block reductions (wave32)
// ---------------------------------------------------------------------------
__device__ __forceinline__ float blockReduce(float v, float* red, bool isMax) {
#pragma unroll
    for (int o = 16; o; o >>= 1) {
        float t = __shfl_xor(v, o, 32);
        v = isMax ? fmaxf(v, t) : (v + t);
    }
    int lane = threadIdx.x & 31, wave = threadIdx.x >> 5;
    if (lane == 0) red[wave] = v;
    __syncthreads();
    if (threadIdx.x == 0) {
        float s = red[0];
        for (int i = 1; i < 8; ++i) s = isMax ? fmaxf(s, red[i]) : (s + red[i]);
        red[0] = s;
    }
    __syncthreads();
    v = red[0];
    __syncthreads();
    return v;
}

// in-place row softmax; one block per row of n elements
__global__ __launch_bounds__(256) void softmax_rows(float* __restrict__ P, int n) {
    __shared__ float red[8];
    const int tid = threadIdx.x;
    float* p = P + (long long)blockIdx.x * n;
    float m = -3.4e38f;
    for (int i = tid; i < n; i += 256) m = fmaxf(m, p[i]);
    m = blockReduce(m, red, true);
    float s = 0.f;
    for (int i = tid; i < n; i += 256) {
        float e = __expf(p[i] - m);
        p[i] = e;
        s += e;
    }
    s = blockReduce(s, red, false);
    float inv = 1.0f / s;
    for (int i = tid; i < n; i += 256) p[i] *= inv;
}

// ---------------------------------------------------------------------------
// out = LayerNorm(x + r) * g + b ; one block per 1024-wide row
// ---------------------------------------------------------------------------
__global__ __launch_bounds__(256) void add_ln_kernel(
    const float* __restrict__ x, const float* __restrict__ r,
    const float* __restrict__ g, const float* __restrict__ b,
    float* __restrict__ o32, bf16* __restrict__ o16)
{
    __shared__ float red[8];
    const int tid = threadIdx.x;
    const long long base = (long long)blockIdx.x * D_;
    float v[4], s = 0.f, s2 = 0.f;
#pragma unroll
    for (int j = 0; j < 4; ++j) {
        int c = tid * 4 + j;
        float t = x[base + c] + r[base + c];
        v[j] = t; s += t; s2 += t * t;
    }
    s  = blockReduce(s, red, false);
    s2 = blockReduce(s2, red, false);
    float mean = s * (1.0f / D_);
    float var  = s2 * (1.0f / D_) - mean * mean;
    float rstd = rsqrtf(var + 1e-5f);
#pragma unroll
    for (int j = 0; j < 4; ++j) {
        int c = tid * 4 + j;
        float y = (v[j] - mean) * rstd * g[c] + b[c];
        o32[base + c] = y;
        if (o16) o16[base + c] = (bf16)y;
    }
}

// ---------------------------------------------------------------------------
extern "C" void kernel_launch(void* const* d_in, const int* in_sizes, int n_in,
                              void* d_out, int out_size, void* d_ws, size_t ws_size,
                              hipStream_t stream)
{
    const float* X    = (const float*)d_in[0];
    const float* Wq   = (const float*)d_in[1];
    const float* bq   = (const float*)d_in[2];
    const float* Wk   = (const float*)d_in[3];
    const float* bk   = (const float*)d_in[4];
    const float* Wv   = (const float*)d_in[5];
    const float* bv   = (const float*)d_in[6];
    const float* Wo   = (const float*)d_in[7];
    const float* bo   = (const float*)d_in[8];
    const float* ln1g = (const float*)d_in[9];
    const float* ln1b = (const float*)d_in[10];
    const float* W1   = (const float*)d_in[11];
    const float* b1   = (const float*)d_in[12];
    const float* W2   = (const float*)d_in[13];
    const float* b2   = (const float*)d_in[14];
    const float* ln2g = (const float*)d_in[15];
    const float* ln2b = (const float*)d_in[16];

    float* Yout  = (float*)d_out;                               // B*L*D f32
    float* Atten = Yout + (long long)M_TOT * D_;                // 32*2048*2048 f32

    char* ws = (char*)d_ws;
    size_t off = 0;
    auto take = [&](size_t bytes) -> void* {
        void* p = ws + off;
        off += (bytes + 255) & ~(size_t)255;
        return p;
    };
    bf16*  Xb   = (bf16*)take((size_t)M_TOT * D_ * 2);   // 16 MB (reused as Ctxb)
    bf16*  Wqb  = (bf16*)take((size_t)D_ * D_ * 2);
    bf16*  Wkb  = (bf16*)take((size_t)D_ * D_ * 2);
    bf16*  Wvb  = (bf16*)take((size_t)D_ * D_ * 2);
    bf16*  Wob  = (bf16*)take((size_t)D_ * D_ * 2);
    bf16*  W1b  = (bf16*)take((size_t)FF_ * D_ * 2);
    bf16*  W2b  = (bf16*)take((size_t)D_ * FF_ * 2);
    bf16*  Qb   = (bf16*)take((size_t)M_TOT * D_ * 2);   // reused as Hb
    bf16*  Kb   = (bf16*)take((size_t)M_TOT * D_ * 2);   // Kb+Vb reused as F1b
    bf16*  Vb   = (bf16*)take((size_t)M_TOT * D_ * 2);
    float* P32  = (float*)take((size_t)M_TOT * D_ * 4);  // proj out, reused as F2
    float* H32  = (float*)take((size_t)M_TOT * D_ * 4);
    bf16*  Ctxb = Xb;
    bf16*  Hb   = Qb;
    bf16*  F1b  = Kb;   // 32 MB spanning Kb..Vb
    float* F2   = P32;
    (void)in_sizes; (void)n_in; (void)out_size; (void)ws_size;

    // ---- pack f32 -> bf16 (inputs + weights) ----
    auto pack = [&](const float* s, bf16* d, int n) {
        f32_to_bf16<<<(n + 255) / 256, 256, 0, stream>>>(s, d, n);
    };
    pack(X,  Xb,  M_TOT * D_);
    pack(Wq, Wqb, D_ * D_);
    pack(Wk, Wkb, D_ * D_);
    pack(Wv, Wvb, D_ * D_);
    pack(Wo, Wob, D_ * D_);
    pack(W1, W1b, FF_ * D_);
    pack(W2, W2b, D_ * FF_);

    const float scale = 0.08838834764831845f;  // DH^-0.5

    // ---- QKV projections: (8192x1024) @ (1024x1024)^T ----
    dim3 gProj(D_ / BN, M_TOT / BM, 1);
    gemm_bt_kernel<<<gProj, 256, 0, stream>>>(Xb, 0, D_, Wqb, 0, D_, D_,
                                              bq, 1.f, 0, nullptr, Qb, 0, D_);
    gemm_bt_kernel<<<gProj, 256, 0, stream>>>(Xb, 0, D_, Wkb, 0, D_, D_,
                                              bk, 1.f, 0, nullptr, Kb, 0, D_);
    gemm_bt_kernel<<<gProj, 256, 0, stream>>>(Xb, 0, D_, Wvb, 0, D_, D_,
                                              bv, 1.f, 0, nullptr, Vb, 0, D_);

    // ---- scores: per group g (raw reshape): S = Qg @ Kg^T * scale -> Atten ----
    dim3 gScore(L_ / BN, L_ / BM, NHEAD);
    gemm_bt_kernel<<<gScore, 256, 0, stream>>>(
        Qb, (long long)L_ * DH_, DH_, Kb, (long long)L_ * DH_, DH_, DH_,
        nullptr, scale, 0, Atten, nullptr, (long long)L_ * L_, L_);

    // ---- softmax rows in place ----
    softmax_rows<<<NHEAD * L_, 256, 0, stream>>>(Atten, L_);

    // ---- context: Ctx = P @ V ----
    dim3 gCtx(1, L_ / BM, NHEAD);
    gemm_ctx_kernel<<<gCtx, 256, 0, stream>>>(
        Atten, (long long)L_ * L_, Vb, (long long)L_ * DH_,
        Ctxb, (long long)L_ * DH_, L_);

    // ---- output projection ----
    gemm_bt_kernel<<<gProj, 256, 0, stream>>>(Ctxb, 0, D_, Wob, 0, D_, D_,
                                              bo, 1.f, 0, P32, nullptr, 0, D_);

    // ---- h = LN(x + proj) ----
    add_ln_kernel<<<M_TOT, 256, 0, stream>>>(X, P32, ln1g, ln1b, H32, Hb);

    // ---- FFN ----
    dim3 gF1(FF_ / BN, M_TOT / BM, 1);
    gemm_bt_kernel<<<gF1, 256, 0, stream>>>(Hb, 0, D_, W1b, 0, D_, D_,
                                            b1, 1.f, 1, nullptr, F1b, 0, FF_);
    dim3 gF2(D_ / BN, M_TOT / BM, 1);
    gemm_bt_kernel<<<gF2, 256, 0, stream>>>(F1b, 0, FF_, W2b, 0, FF_, FF_,
                                            b2, 1.f, 0, F2, nullptr, 0, D_);

    // ---- y = LN(h + f) ----
    add_ln_kernel<<<M_TOT, 256, 0, stream>>>(H32, F2, ln2g, ln2b, Yout, nullptr);
}
